// AdvancedAttentionLayer_43439299232078
// MI455X (gfx1250) — matrix operand order repaired and checked by reading, not compiled, not run on hardware
//
#include <hip/hip_runtime.h>

// ---------------------------------------------------------------------------
// MI455X (gfx1250) attention layer: bf16 WMMA pipeline, fp32 accumulate,
// async-to-LDS (ASYNCcnt) double-buffered data movement.
// B=2, S=2048, D=1024, H=16, DH=64.
// ---------------------------------------------------------------------------

typedef __bf16 bf16_t;
typedef bf16_t v16bf __attribute__((ext_vector_type(16)));
typedef bf16_t v8bf  __attribute__((ext_vector_type(8)));
typedef bf16_t v4bf  __attribute__((ext_vector_type(4)));
typedef float  v8f   __attribute__((ext_vector_type(8)));
typedef int    v4i32 __attribute__((vector_size(16)));   // matches builtin param

#define B_DIM 2
#define S_DIM 2048
#define D_DIM 1024
#define H_DIM 16
#define DH_DIM 64
#define M_ROWS (B_DIM * S_DIM)   // 4096

// ---------------- CDNA5 async copy (global -> LDS, ASYNCcnt) ----------------
__device__ __forceinline__ void async_cp16(const bf16_t* g, bf16_t* l) {
#if __has_builtin(__builtin_amdgcn_global_load_async_to_lds_b128)
  __builtin_amdgcn_global_load_async_to_lds_b128(
      (__attribute__((address_space(1))) v4i32*)g,
      (__attribute__((address_space(3))) v4i32*)l, 0, 0);
#else
  asm volatile("global_load_async_to_lds_b128 %0, %1, off"
               :: "v"((unsigned)(unsigned long long)l),
                  "v"((unsigned long long)g)
               : "memory");
#endif
}

__device__ __forceinline__ void wait_async0() {
#if __has_builtin(__builtin_amdgcn_s_wait_asynccnt)
  __builtin_amdgcn_s_wait_asynccnt(0);
#else
  asm volatile("s_wait_asynccnt 0x0" ::: "memory");
#endif
}

// ---------------- small helpers ----------------
__device__ __forceinline__ v8bf ld8(const bf16_t* p) { return *(const v8bf*)p; }

__device__ __forceinline__ v16bf cmb(v8bf lo, v8bf hi) {
  v16bf r;
#pragma unroll
  for (int i = 0; i < 8; ++i) { r[i] = lo[i]; r[i + 8] = hi[i]; }
  return r;
}

__device__ __forceinline__ v8f v8f_zero() {
  v8f z;
#pragma unroll
  for (int i = 0; i < 8; ++i) z[i] = 0.0f;
  return z;
}

__device__ __forceinline__ v8f wmma_bf16(v16bf a, v16bf b, v8f c) {
  return __builtin_amdgcn_wmma_f32_16x16x32_bf16(false, a, false, b,
                                                 (short)0, c, false, false);
}

// reduce across the 16 lanes that share one matrix row (wave32 halves)
__device__ __forceinline__ float redmax16(float v) {
  v = fmaxf(v, __shfl_xor(v, 1, 32));
  v = fmaxf(v, __shfl_xor(v, 2, 32));
  v = fmaxf(v, __shfl_xor(v, 4, 32));
  v = fmaxf(v, __shfl_xor(v, 8, 32));
  return v;
}
__device__ __forceinline__ float redsum16(float v) {
  v += __shfl_xor(v, 1, 32);
  v += __shfl_xor(v, 2, 32);
  v += __shfl_xor(v, 4, 32);
  v += __shfl_xor(v, 8, 32);
  return v;
}

// ---------------- stage 1a: fp32 -> bf16 (activations) ----------------
__global__ __launch_bounds__(256) void k_f32_to_bf16(const float* __restrict__ src,
                                                     bf16_t* __restrict__ dst, int n) {
  int i = (blockIdx.x * 256 + threadIdx.x) * 4;
  if (i < n) {
    float4 v = *(const float4*)(src + i);
    v4bf o;
    o[0] = (bf16_t)v.x; o[1] = (bf16_t)v.y; o[2] = (bf16_t)v.z; o[3] = (bf16_t)v.w;
    *(v4bf*)(dst + i) = o;
  }
}

// ---------------- stage 1b: W[k][n] (f32) -> Wt[n][k] (bf16), LDS tiled ------
__global__ __launch_bounds__(256) void k_transpose_to_bf16(const float* __restrict__ W,
                                                           bf16_t* __restrict__ Wt) {
  __shared__ float tile[32][33];
  const int tx = threadIdx.x, ty = threadIdx.y;
#pragma unroll
  for (int i = 0; i < 4; ++i) {
    int k = blockIdx.y * 32 + ty + i * 8;
    int n = blockIdx.x * 32 + tx;
    tile[ty + i * 8][tx] = W[(size_t)k * D_DIM + n];
  }
  __syncthreads();
#pragma unroll
  for (int i = 0; i < 4; ++i) {
    int n2 = blockIdx.x * 32 + ty + i * 8;
    int k2 = blockIdx.y * 32 + tx;
    Wt[(size_t)n2 * D_DIM + k2] = (bf16_t)tile[tx][ty + i * 8];
  }
}

// ---------------- stage 2/5: bf16 WMMA GEMM, async double-buffered ----------
// C[M,N] = A[M,K] * Bt[N,K]^T (+ bias[col] + add2[col])
// mode 0: store bf16 to [b,h,s,dh]  (Q with emotion add2, K)
// mode 2: store bf16 to [b,h,dh,s]  (V transposed for attention B-frags)
// mode 3: store f32  to [row, col]  (final output)
__global__ __launch_bounds__(256) void k_gemm_bf16(const bf16_t* __restrict__ A,
                                                   const bf16_t* __restrict__ Bt,
                                                   const float* __restrict__ bias,
                                                   const float* __restrict__ add2,
                                                   void* __restrict__ out, int mode) {
  __shared__ __align__(16) bf16_t As[2][64 * 32];    // 2 x 4 KB
  __shared__ __align__(16) bf16_t Bs[2][128 * 32];   // 2 x 8 KB
  const int K = D_DIM;
  const int tid  = threadIdx.x;
  const int lane = tid & 31;
  const int wave = tid >> 5;
  const int wm = wave >> 2;               // 0..1  (M sub-tile of 32)
  const int wn = wave & 3;                // 0..3  (N sub-tile of 32)
  const int mbase = blockIdx.y * 64;
  const int nbase = blockIdx.x * 128;
  const int l15 = lane & 15;
  const int hl  = (lane >> 4) & 1;
  const int bk  = hl * 8;                 // per-lane K base (A/B frag layout)

  v8f acc[2][2];
#pragma unroll
  for (int i = 0; i < 2; ++i)
#pragma unroll
    for (int j = 0; j < 2; ++j) acc[i][j] = v8f_zero();

  // tile-fill ownership: 3 x 16B async copies per thread per K-step
  const int arow = tid >> 2, ach = (tid & 3) * 8;            // A: 64x32
  const int br0 = tid >> 2, bo0 = (tid & 3) * 8;             // B: 128x32 (two chunks)
  const int br1 = (tid + 256) >> 2, bo1 = ((tid + 256) & 3) * 8;

  auto fill = [&](int buf, int kk) {
    async_cp16(&A [(size_t)(mbase + arow) * K + kk + ach], &As[buf][arow * 32 + ach]);
    async_cp16(&Bt[(size_t)(nbase + br0)  * K + kk + bo0], &Bs[buf][br0  * 32 + bo0]);
    async_cp16(&Bt[(size_t)(nbase + br1)  * K + kk + bo1], &Bs[buf][br1  * 32 + bo1]);
  };

  fill(0, 0);
  const int T = K / 32;
  for (int t = 0; t < T; ++t) {
    const int cur = t & 1;
    wait_async0();          // our async fills for buf[cur] have landed in LDS
    __syncthreads();        // everyone's fills landed -> tile readable
    if (t + 1 < T) fill(cur ^ 1, (t + 1) * 32);   // overlap next copy with math

    v16bf a[2], b[2];
#pragma unroll
    for (int i = 0; i < 2; ++i) {
      int r = (wm * 32 + i * 16 + l15) * 32;
      a[i] = cmb(ld8(&As[cur][r + bk]), ld8(&As[cur][r + 16 + bk]));
    }
#pragma unroll
    for (int j = 0; j < 2; ++j) {
      int r = (wn * 32 + j * 16 + l15) * 32;
      b[j] = cmb(ld8(&Bs[cur][r + bk]), ld8(&Bs[cur][r + 16 + bk]));
    }
#pragma unroll
    for (int i = 0; i < 2; ++i)
#pragma unroll
      for (int j = 0; j < 2; ++j) acc[i][j] = wmma_bf16(a[i], b[j], acc[i][j]);
    // no trailing barrier needed: frag ds_loads are dscnt-waited before the
    // WMMAs above, which precede this wave's arrival at the next barrier.
  }

  // epilogue
  bf16_t* bout = (bf16_t*)out;
  float*  fout = (float*)out;
#pragma unroll
  for (int i = 0; i < 2; ++i) {
#pragma unroll
    for (int j = 0; j < 2; ++j) {
      const int col = nbase + wn * 32 + j * 16 + l15;
      const float extra = (bias ? bias[col] : 0.0f) + (add2 ? add2[col] : 0.0f);
      const int hh = col >> 6, dh = col & 63;
#pragma unroll
      for (int r = 0; r < 8; ++r) {
        const int row = mbase + wm * 32 + i * 16 + r + hl * 8;
        const float v = acc[i][j][r] + extra;
        if (mode == 3) {
          fout[(size_t)row * D_DIM + col] = v;
        } else {
          const int bb = row >> 11, s = row & 2047;
          if (mode == 0) {  // [b,h,s,dh]
            bout[(((size_t)(bb * H_DIM + hh)) * S_DIM + s) * DH_DIM + dh] = (bf16_t)v;
          } else {          // mode 2: [b,h,dh,s]
            bout[(((size_t)(bb * H_DIM + hh)) * DH_DIM + dh) * S_DIM + s] = (bf16_t)v;
          }
        }
      }
    }
  }
}

// ---------------- stage 3: flash attention ----------------------------------
// 4 waves/block; each wave owns a 16-row Q tile. K/V blocks (32 keys) are
// staged in LDS via async copies, double-buffered, and SHARED by all 4 waves
// (they all work on the same (b,h)).
__global__ __launch_bounds__(128) void k_flash_attn(const bf16_t* __restrict__ Qb,
                                                    const bf16_t* __restrict__ Kb,
                                                    const bf16_t* __restrict__ Vt,
                                                    const int* __restrict__ amask,
                                                    bf16_t* __restrict__ Ob) {
  __shared__ __align__(16) bf16_t Ks[2][32 * 64];   // 2 x 4 KB, [key][dh]
  __shared__ __align__(16) bf16_t Vs[2][64 * 32];   // 2 x 4 KB, [dh][key]
  __shared__ __align__(16) bf16_t Plds[4][16 * 32]; // per-wave P staging (C->A)
  const int tid  = threadIdx.x;
  const int lane = tid & 31;
  const int wave = tid >> 5;
  const int bh = blockIdx.y;           // b*H + h
  const int b  = bh >> 4;
  const int h  = bh & 15;
  const int qbase = blockIdx.x * 64 + wave * 16;
  const int l15 = lane & 15;
  const int hl  = (lane >> 4) & 1;
  const int bk  = hl * 8;

  // Q A-fragments (dh 0..31 and 32..63), loaded once
  const bf16_t* qrow = Qb + ((size_t)bh * S_DIM + qbase + l15) * DH_DIM;
  const v16bf qa0 = cmb(ld8(qrow + bk),      ld8(qrow + 16 + bk));
  const v16bf qa1 = cmb(ld8(qrow + 32 + bk), ld8(qrow + 48 + bk));

  v8f o0 = v8f_zero(), o1 = v8f_zero(), o2 = v8f_zero(), o3 = v8f_zero();
  v8f mrow, lrow;
#pragma unroll
  for (int r = 0; r < 8; ++r) { mrow[r] = -1e30f; lrow[r] = 0.0f; }

  const int* mk = amask + b * S_DIM;
  const bf16_t* kbase = Kb + (size_t)bh * S_DIM * DH_DIM;
  const bf16_t* vbase = Vt + (size_t)bh * DH_DIM * S_DIM;
  bf16_t* plds = &Plds[wave][0];

  // K tile 4KB = 256 16B chunks; V tile 4KB = 256 chunks; 128 threads -> 4 each
  const int kc0 = tid,       kr0 = kc0 >> 3, ko0 = (kc0 & 7) * 8;
  const int kc1 = tid + 128, kr1 = kc1 >> 3, ko1 = (kc1 & 7) * 8;
  const int vc0 = tid,       vr0 = vc0 >> 2, vo0 = (vc0 & 3) * 8;
  const int vc1 = tid + 128, vr1 = vc1 >> 2, vo1 = (vc1 & 3) * 8;

  auto fillkv = [&](int buf, int key0) {
    async_cp16(kbase + (size_t)(key0 + kr0) * DH_DIM + ko0, &Ks[buf][kr0 * 64 + ko0]);
    async_cp16(kbase + (size_t)(key0 + kr1) * DH_DIM + ko1, &Ks[buf][kr1 * 64 + ko1]);
    async_cp16(vbase + (size_t)vr0 * S_DIM + key0 + vo0,    &Vs[buf][vr0 * 32 + vo0]);
    async_cp16(vbase + (size_t)vr1 * S_DIM + key0 + vo1,    &Vs[buf][vr1 * 32 + vo1]);
  };

  fillkv(0, 0);
  for (int blk = 0; blk < S_DIM / 32; ++blk) {
    const int cur = blk & 1;
    const int key0 = blk * 32;
    wait_async0();
    __syncthreads();
    if (blk + 1 < S_DIM / 32) fillkv(cur ^ 1, (blk + 1) * 32);

    // ---- scores: S = Q(16x64) . K^T(64x32), from shared K tile ----
    v8f sc[2];
#pragma unroll
    for (int sub = 0; sub < 2; ++sub) {
      const int kr = (sub * 16 + l15) * 64;
      const v16bf kb0 = cmb(ld8(&Ks[cur][kr + bk]),      ld8(&Ks[cur][kr + 16 + bk]));
      const v16bf kb1 = cmb(ld8(&Ks[cur][kr + 32 + bk]), ld8(&Ks[cur][kr + 48 + bk]));
      v8f c = v8f_zero();
      c = wmma_bf16(qa0, kb0, c);
      c = wmma_bf16(qa1, kb1, c);
      const bool dead = (mk[key0 + sub * 16 + l15] == 0);  // key-padding mask
#pragma unroll
      for (int r = 0; r < 8; ++r) sc[sub][r] = dead ? -1e30f : c[r] * 0.125f;
    }

    // ---- online softmax (row = accumulator register, 16-lane reductions) ----
    v8f alpha;
#pragma unroll
    for (int r = 0; r < 8; ++r) {
      float rm = redmax16(fmaxf(sc[0][r], sc[1][r]));
      float nm = fmaxf(mrow[r], rm);
      alpha[r] = __expf(mrow[r] - nm);
      mrow[r] = nm;
      float p0 = __expf(sc[0][r] - nm);
      float p1 = __expf(sc[1][r] - nm);
      sc[0][r] = p0; sc[1][r] = p1;
      lrow[r] = lrow[r] * alpha[r] + redsum16(p0 + p1);
    }
#pragma unroll
    for (int r = 0; r < 8; ++r) {
      o0[r] *= alpha[r]; o1[r] *= alpha[r]; o2[r] *= alpha[r]; o3[r] *= alpha[r];
    }

    // ---- P: C-layout -> A-layout via per-wave LDS tile [16 rows][32 keys] ----
#pragma unroll
    for (int r = 0; r < 8; ++r) {
      const int m = r + hl * 8;
      plds[m * 32 + l15]      = (bf16_t)sc[0][r];
      plds[m * 32 + 16 + l15] = (bf16_t)sc[1][r];
    }
    asm volatile("s_wait_dscnt 0x0" ::: "memory");  // same-wave DS RAW guard
    const int prow = l15 * 32;
    const v16bf pa = cmb(ld8(&plds[prow + bk]), ld8(&plds[prow + 16 + bk]));

    // ---- O += P(16x32) . V(32x64), from shared V tile ([dh][key]) ----
#pragma unroll
    for (int c4 = 0; c4 < 4; ++c4) {
      const int vr = (c4 * 16 + l15) * 32;
      const v16bf vb = cmb(ld8(&Vs[cur][vr + bk]), ld8(&Vs[cur][vr + 16 + bk]));
      if (c4 == 0) o0 = wmma_bf16(pa, vb, o0);
      else if (c4 == 1) o1 = wmma_bf16(pa, vb, o1);
      else if (c4 == 2) o2 = wmma_bf16(pa, vb, o2);
      else o3 = wmma_bf16(pa, vb, o3);
    }
  }

  // finalize: divide by softmax denom, store bf16 [row = b*S+s][col = h*64+dh]
#pragma unroll
  for (int r = 0; r < 8; ++r) {
    const float inv = 1.0f / lrow[r];
    const int srow = qbase + r + hl * 8;
    const size_t base = ((size_t)b * S_DIM + srow) * D_DIM + h * DH_DIM;
    Ob[base + 0  + l15] = (bf16_t)(o0[r] * inv);
    Ob[base + 16 + l15] = (bf16_t)(o1[r] * inv);
    Ob[base + 32 + l15] = (bf16_t)(o2[r] * inv);
    Ob[base + 48 + l15] = (bf16_t)(o3[r] * inv);
  }
}

// ---------------------------------------------------------------------------
extern "C" void kernel_launch(void* const* d_in, const int* in_sizes, int n_in,
                              void* d_out, int out_size, void* d_ws, size_t ws_size,
                              hipStream_t stream) {
  const float* hidden = (const float*)d_in[0];
  const int*   amask  = (const int*)d_in[1];
  const float* Wq = (const float*)d_in[2];
  const float* bq = (const float*)d_in[3];
  const float* Wk = (const float*)d_in[4];
  const float* bk = (const float*)d_in[5];
  const float* Wv = (const float*)d_in[6];
  const float* bv = (const float*)d_in[7];
  const float* emo = (const float*)d_in[8];   // [H,DH] flat == per-column vector
  const float* Wo = (const float*)d_in[9];
  const float* bo = (const float*)d_in[10];

  char* ws = (char*)d_ws;
  const size_t MB = 1u << 20;
  bf16_t* Xb  = (bf16_t*)(ws + 0);          // 8 MB  [4096][1024]
  bf16_t* WqT = (bf16_t*)(ws + 8 * MB);     // 2 MB  [N][K]
  bf16_t* WkT = (bf16_t*)(ws + 10 * MB);
  bf16_t* WvT = (bf16_t*)(ws + 12 * MB);
  bf16_t* WoT = (bf16_t*)(ws + 14 * MB);
  bf16_t* Qb  = (bf16_t*)(ws + 16 * MB);    // 8 MB  [b,h,s,dh]
  bf16_t* Kb  = (bf16_t*)(ws + 24 * MB);    // 8 MB  [b,h,s,dh]
  bf16_t* Vt  = (bf16_t*)(ws + 32 * MB);    // 8 MB  [b,h,dh,s]
  bf16_t* Ob  = (bf16_t*)(ws + 40 * MB);    // 8 MB  [4096][1024]

  // stage 1: precision conversion + weight transposes
  const int nX = M_ROWS * D_DIM;
  k_f32_to_bf16<<<nX / (256 * 4), 256, 0, stream>>>(hidden, Xb, nX);
  dim3 tb(32, 8), tg(D_DIM / 32, D_DIM / 32);
  k_transpose_to_bf16<<<tg, tb, 0, stream>>>(Wq, WqT);
  k_transpose_to_bf16<<<tg, tb, 0, stream>>>(Wk, WkT);
  k_transpose_to_bf16<<<tg, tb, 0, stream>>>(Wv, WvT);
  k_transpose_to_bf16<<<tg, tb, 0, stream>>>(Wo, WoT);

  // stage 2: fused QKV projections (Q gets emotion bias in epilogue)
  dim3 gg(D_DIM / 128, M_ROWS / 64);
  k_gemm_bf16<<<gg, 256, 0, stream>>>(Xb, WqT, bq, emo, (void*)Qb, 0);
  k_gemm_bf16<<<gg, 256, 0, stream>>>(Xb, WkT, bk, nullptr, (void*)Kb, 0);
  k_gemm_bf16<<<gg, 256, 0, stream>>>(Xb, WvT, bv, nullptr, (void*)Vt, 2);

  // stage 3: flash attention — grid (q-chunks of 64, B*H)
  dim3 ag(S_DIM / 64, B_DIM * H_DIM);
  k_flash_attn<<<ag, 128, 0, stream>>>(Qb, Kb, Vt, amask, Ob);

  // stage 4: output projection -> fp32 d_out
  k_gemm_bf16<<<gg, 256, 0, stream>>>(Ob, WoT, bo, nullptr, d_out, 3);
}
